// HebbianMemory_48739288875016
// MI455X (gfx1250) — matrix operand order repaired and checked by reading, not compiled, not run on hardware
//
#include <hip/hip_runtime.h>

// ---------------------------------------------------------------------------
// CDNA5 (gfx1250) HebbianMemory: chunked linear attention, bf16 WMMA pipeline
//   - all WMMA operand fragments are contiguous 16B LDS loads (pre-swizzled /
//     pre-transposed operand layouts)
//   - async global->LDS staging (ASYNCcnt) with double buffering / chunk
//     prefetch where the builtin is available
// ---------------------------------------------------------------------------

typedef __bf16 bf16_t;
typedef __attribute__((ext_vector_type(16))) bf16_t v16bf;
typedef __attribute__((ext_vector_type(8)))  float  v8f;

#define D_MODEL 512
#define L_SEQ   2048
#define BATCH   2
#define HEADS   8
#define EDIM    64
#define ROWS    (BATCH * L_SEQ)   // 4096
#define N_QKV   (3 * D_MODEL)     // 1536
#define CHUNK   64
#define NCHUNK  (L_SEQ / CHUNK)   // 32

static __device__ __forceinline__ float  bf2f(bf16_t b) { return (float)b; }
static __device__ __forceinline__ bf16_t f2bf(float f)  { return (bf16_t)f; }

// ---- async global->LDS (CDNA5 GLOBAL_LOAD_ASYNC_TO_LDS_*) ------------------
#if __has_builtin(__builtin_amdgcn_global_load_async_to_lds_b128) && \
    __has_builtin(__builtin_amdgcn_global_load_async_to_lds_b64)
#define HAVE_ASYNC_LDS 1
typedef int v4i_ __attribute__((vector_size(16)));
typedef int v2i_ __attribute__((vector_size(8)));
typedef __attribute__((address_space(1))) v4i_ as1_v4i;
typedef __attribute__((address_space(3))) v4i_ as3_v4i;
typedef __attribute__((address_space(1))) v2i_ as1_v2i;
typedef __attribute__((address_space(3))) v2i_ as3_v2i;
#define AWAIT_ASYNC(n) asm volatile("s_wait_asynccnt " #n ::: "memory")
static __device__ __forceinline__ void async_cp16(const void* g, void* l) {
  __builtin_amdgcn_global_load_async_to_lds_b128((as1_v4i*)g, (as3_v4i*)l, 0, 0);
}
static __device__ __forceinline__ void async_cp8(const void* g, void* l) {
  __builtin_amdgcn_global_load_async_to_lds_b64((as1_v2i*)g, (as3_v2i*)l, 0, 0);
}
#else
#define HAVE_ASYNC_LDS 0
#define AWAIT_ASYNC(n) do {} while (0)
static __device__ __forceinline__ void async_cp16(const void* g, void* l) {
  *(uint4*)l = *(const uint4*)g;
}
static __device__ __forceinline__ void async_cp8(const void* g, void* l) {
  *(uint2*)l = *(const uint2*)g;
}
#endif

static __device__ __forceinline__ v8f wmma_bf16(v16bf a, v16bf b, v8f c) {
  return __builtin_amdgcn_wmma_f32_16x16x32_bf16(false, a, false, b, (short)0, c,
                                                 false, false);
}

// ---- fragment loaders: all contiguous 16B LDS reads ------------------------
// A operand (16x32 tile of a row-major [M][K] source):
//   lane(m,h): elements A[m][k0+8h .. +7] and A[m][k0+16+8h .. +7]
static __device__ __forceinline__ v16bf frag_row(const bf16_t* s, int ld,
                                                 int row0, int k0, int lane) {
  int m = lane & 15, h = lane >> 4;
  const bf16_t* p = s + (row0 + m) * ld + k0 + 8 * h;
  v16bf r;
  ((uint4*)&r)[0] = *(const uint4*)(p);
  ((uint4*)&r)[1] = *(const uint4*)(p + 16);
  return r;
}
// B operand (32x16 tile) taken from a PRE-TRANSPOSED source sT[N][K]:
//   lane(n,h): elements sT[col0+n][k0+8h .. +7] and sT[col0+n][k0+16+8h .. +7]
static __device__ __forceinline__ v16bf frag_colT(const bf16_t* sT, int ld,
                                                  int k0, int col0, int lane) {
  int n = lane & 15, h = lane >> 4;
  const bf16_t* p = sT + (col0 + n) * ld + k0 + 8 * h;
  v16bf r;
  ((uint4*)&r)[0] = *(const uint4*)(p);
  ((uint4*)&r)[1] = *(const uint4*)(p + 16);
  return r;
}

// --- Kernel 0a: f32 -> bf16 convert ----------------------------------------
__global__ __launch_bounds__(256) void convert_kernel(const float* __restrict__ src,
                                                      bf16_t* __restrict__ dst,
                                                      int n) {
  int i = blockIdx.x * 256 + threadIdx.x;
  if (i < n) dst[i] = f2bf(src[i]);
}
// --- Kernel 0b: f32 [R][C] -> bf16 transposed [C][R] ------------------------
__global__ __launch_bounds__(256) void convert_t_kernel(const float* __restrict__ src,
                                                        bf16_t* __restrict__ dst,
                                                        int R, int C) {
  int i = blockIdx.x * 256 + threadIdx.x;
  if (i < R * C) {
    int r = i / C, c = i - r * C;
    dst[(size_t)c * R + r] = f2bf(src[i]);
  }
}

// --- Kernel 1: QKV GEMM, fused bias + elu+1 + scaling + decay --------------
// Block: 64 rows x 32 cols of [4096 x 1536]; 8 waves, one 16x16 tile each.
// Double-buffered async LDS staging; weights pre-transposed [N][K].
__global__ __launch_bounds__(256) void qkv_gemm_kernel(
    const bf16_t* __restrict__ Xb, const bf16_t* __restrict__ WbT,
    const float* __restrict__ bias, const float* __restrict__ decay_logits,
    bf16_t* __restrict__ qws, bf16_t* __restrict__ kws, bf16_t* __restrict__ vws) {
  __shared__ __align__(16) bf16_t As[2][64 * 32];
  __shared__ __align__(16) bf16_t BsT[2][32 * 32];
  const int tid = threadIdx.x;
  const int lane = tid & 31, wave = tid >> 5;
  const int row_blk = blockIdx.y * 64;
  const int col_blk = blockIdx.x * 32;
  const int tr = wave & 3, tc = wave >> 2;

  const int ra = tid >> 2, sa = tid & 3;   // A: 64x32, 8 bf16/thread
  const int rb = tid >> 3, sb = tid & 7;   // B: 32x32, 4 bf16/thread
  auto issue = [&](int kk, int nb) {
    async_cp16(Xb + (size_t)(row_blk + ra) * D_MODEL + kk + sa * 8,
               &As[nb][ra * 32 + sa * 8]);
    async_cp8(WbT + (size_t)(col_blk + rb) * D_MODEL + kk + sb * 4,
              &BsT[nb][rb * 32 + sb * 4]);
  };

  issue(0, 0);
  v8f acc = {};
  for (int i = 0; i < D_MODEL / 32; ++i) {
    int buf = i & 1;
    if (i + 1 < D_MODEL / 32) {
      issue((i + 1) * 32, buf ^ 1);
      AWAIT_ASYNC(0x2);
    } else {
      AWAIT_ASYNC(0x0);
    }
    __syncthreads();
    acc = wmma_bf16(frag_row(As[buf], 32, tr * 16, 0, lane),
                    frag_colT(BsT[buf], 32, 0, tc * 16, lane), acc);
    __syncthreads();
  }

  const int n = lane & 15, hh = lane >> 4;
#pragma unroll
  for (int r = 0; r < 8; ++r) {
    int row = row_blk + tr * 16 + r + 8 * hh;
    int col = col_blk + tc * 16 + n;
    float v = acc[r] + bias[col];
    int part = col / D_MODEL;  // 0=q, 1=k, 2=v
    int cc = col - part * D_MODEL;
    int head = cc >> 6;
    int t = row & (L_SEQ - 1);
    float lam = 0.99f + 0.01f * (1.0f / (1.0f + __expf(-decay_logits[head])));
    float ll = __logf(lam);
    size_t o = (size_t)row * D_MODEL + cc;
    if (part == 0) {
      float q = (v > 0.0f ? v + 1.0f : __expf(v)) * 0.125f;  // /sqrt(E)
      qws[o] = f2bf(q * __expf(ll * (float)t));               // q * lam^t
    } else if (part == 1) {
      float k = (v > 0.0f ? v + 1.0f : __expf(v));
      kws[o] = f2bf(k * __expf(-ll * (float)t));              // k * lam^-t
    } else {
      vws[o] = f2bf(v);
    }
  }
}

// --- Kernel 2: chunked linear-attention scan -------------------------------
// One block per (b, head); 8 waves; 32 sequential chunks of 64 tokens.
// Next chunk's Q/K/V are async-prefetched while the current chunk computes.
__global__ __launch_bounds__(256) void scan_kernel(
    const bf16_t* __restrict__ qws, const bf16_t* __restrict__ kws,
    const bf16_t* __restrict__ vws, bf16_t* __restrict__ attn) {
  __shared__ __align__(16) bf16_t Qbuf[2][CHUNK * EDIM];
  __shared__ __align__(16) bf16_t Kbuf[2][CHUNK * EDIM];
  __shared__ __align__(16) bf16_t Vbuf[2][CHUNK * EDIM];
  __shared__ __align__(16) bf16_t KsT[EDIM * CHUNK];   // [e][s]
  __shared__ __align__(16) bf16_t VsT[EDIM * CHUNK];   // [f][s]
  __shared__ __align__(16) bf16_t Ss[CHUNK * CHUNK];   // [t][s]
  __shared__ __align__(16) bf16_t MbfT[EDIM * EDIM];   // [f][e]
  __shared__ float Mst[EDIM * EDIM];                   // [e][f]
  __shared__ float ksum[EDIM];
  __shared__ float den[CHUNK];

  const int tid = threadIdx.x;
  const int lane = tid & 31, wave = tid >> 5;
  const int bh = blockIdx.x;
  const int b = bh >> 3, head = bh & 7;
  const int n = lane & 15, hh = lane >> 4;

  for (int i = tid; i < EDIM * EDIM; i += 256) {
    Mst[i] = 0.0f;
    MbfT[i] = f2bf(0.0f);
  }
  if (tid < EDIM) ksum[tid] = 0.0f;
  __syncthreads();

  auto issue_chunk = [&](int c, int nb) {
#pragma unroll
    for (int it = 0; it < 2; ++it) {
      int idx = tid + 256 * it;
      int r = idx >> 3, seg = idx & 7;
      size_t g = (size_t)(b * L_SEQ + c * CHUNK + r) * D_MODEL + head * EDIM + seg * 8;
      int l = r * EDIM + seg * 8;
      async_cp16(qws + g, &Qbuf[nb][l]);
      async_cp16(kws + g, &Kbuf[nb][l]);
      async_cp16(vws + g, &Vbuf[nb][l]);
    }
  };

  issue_chunk(0, 0);
  for (int c = 0; c < NCHUNK; ++c) {
    const int buf = c & 1;
    if (c + 1 < NCHUNK) {
      issue_chunk(c + 1, buf ^ 1);
      AWAIT_ASYNC(0x6);   // allow next chunk's 6 async ops to stay in flight
    } else {
      AWAIT_ASYNC(0x0);
    }
    __syncthreads();  // chunk c data resident in LDS for all waves

    const bf16_t* Qs = Qbuf[buf];
    const bf16_t* Ks = Kbuf[buf];   // [s][e]: already B-layout for Q.K^T

    // transpose K,V into [e][s]/[f][s]; zero den
#pragma unroll
    for (int it = 0; it < 2; ++it) {
      int idx = tid + 256 * it;
      int r = idx >> 3, seg = idx & 7;
      uint4 kq = *(const uint4*)(Ks + r * EDIM + seg * 8);
      uint4 vq = *(const uint4*)(Vbuf[buf] + r * EDIM + seg * 8);
      const bf16_t* tk = (const bf16_t*)&kq;
      const bf16_t* tv = (const bf16_t*)&vq;
#pragma unroll
      for (int j = 0; j < 8; ++j) {
        KsT[(seg * 8 + j) * CHUNK + r] = tk[j];
        VsT[(seg * 8 + j) * CHUNK + r] = tv[j];
      }
    }
    if (tid < CHUNK) den[tid] = 0.0f;
    __syncthreads();

    // den_inter[t] = Q[t,:] . ksum
    {
      int t = tid >> 2, q = tid & 3;
      float s = 0.0f;
#pragma unroll
      for (int j = 0; j < 16; ++j)
        s += bf2f(Qs[t * EDIM + q * 16 + j]) * ksum[q * 16 + j];
      atomicAdd(&den[t], s);
    }
    // S = Q.K^T (causal mask, row-sums into den)
#pragma unroll
    for (int i = 0; i < 2; ++i) {
      int tile = wave * 2 + i, tr = tile >> 2, tc = tile & 3;
      v8f acc = {};
#pragma unroll
      for (int ke = 0; ke < EDIM; ke += 32)
        acc = wmma_bf16(frag_row(Qs, EDIM, tr * 16, ke, lane),
                        frag_colT(Ks, EDIM, ke, tc * 16, lane), acc);
#pragma unroll
      for (int r = 0; r < 8; ++r) {
        int t = tr * 16 + r + 8 * hh, s = tc * 16 + n;
        float v = (s <= t) ? acc[r] : 0.0f;
        Ss[t * CHUNK + s] = f2bf(v);
        atomicAdd(&den[t], v);
      }
    }
    __syncthreads();

    // O = S.V + Q.M ; divide by den; store
#pragma unroll
    for (int i = 0; i < 2; ++i) {
      int tile = wave * 2 + i, tr = tile >> 2, tc = tile & 3;
      v8f acc = {};
#pragma unroll
      for (int ke = 0; ke < EDIM; ke += 32) {
        acc = wmma_bf16(frag_row(Ss, CHUNK, tr * 16, ke, lane),
                        frag_colT(VsT, CHUNK, ke, tc * 16, lane), acc);
        acc = wmma_bf16(frag_row(Qs, EDIM, tr * 16, ke, lane),
                        frag_colT(MbfT, EDIM, ke, tc * 16, lane), acc);
      }
#pragma unroll
      for (int r = 0; r < 8; ++r) {
        int tl = tr * 16 + r + 8 * hh, f = tc * 16 + n;
        float o = acc[r] / (den[tl] + 1e-6f);
        size_t g = (size_t)(b * L_SEQ + c * CHUNK + tl) * D_MODEL + head * EDIM + f;
        attn[g] = f2bf(o);
      }
    }
    __syncthreads();

    // state update: M += K^T.V ; ksum += colsum(K)
#pragma unroll
    for (int i = 0; i < 2; ++i) {
      int tile = wave * 2 + i, tr = tile >> 2, tc = tile & 3;
      v8f acc = {};
#pragma unroll
      for (int ks = 0; ks < CHUNK; ks += 32)
        acc = wmma_bf16(frag_row(KsT, CHUNK, tr * 16, ks, lane),
                        frag_colT(VsT, CHUNK, ks, tc * 16, lane), acc);
#pragma unroll
      for (int r = 0; r < 8; ++r) {
        int e = tr * 16 + r + 8 * hh, f = tc * 16 + n;
        float nv = Mst[e * EDIM + f] + acc[r];
        Mst[e * EDIM + f] = nv;
        MbfT[f * EDIM + e] = f2bf(nv);
      }
    }
    {
      int e = tid >> 2, q = tid & 3;
      float s = 0.0f;
#pragma unroll
      for (int j = 0; j < 16; ++j) s += bf2f(KsT[e * CHUNK + q * 16 + j]);
      atomicAdd(&ksum[e], s);
    }
    __syncthreads();
  }
}

// --- Kernel 3: output projection GEMM (+bias) ------------------------------
__global__ __launch_bounds__(256) void out_gemm_kernel(
    const bf16_t* __restrict__ Ab, const bf16_t* __restrict__ WbT,
    const float* __restrict__ bias, float* __restrict__ out) {
  __shared__ __align__(16) bf16_t As[2][64 * 32];
  __shared__ __align__(16) bf16_t BsT[2][32 * 32];
  const int tid = threadIdx.x;
  const int lane = tid & 31, wave = tid >> 5;
  const int row_blk = blockIdx.y * 64;
  const int col_blk = blockIdx.x * 32;
  const int tr = wave & 3, tc = wave >> 2;

  const int ra = tid >> 2, sa = tid & 3;
  const int rb = tid >> 3, sb = tid & 7;
  auto issue = [&](int kk, int nb) {
    async_cp16(Ab + (size_t)(row_blk + ra) * D_MODEL + kk + sa * 8,
               &As[nb][ra * 32 + sa * 8]);
    async_cp8(WbT + (size_t)(col_blk + rb) * D_MODEL + kk + sb * 4,
              &BsT[nb][rb * 32 + sb * 4]);
  };

  issue(0, 0);
  v8f acc = {};
  for (int i = 0; i < D_MODEL / 32; ++i) {
    int buf = i & 1;
    if (i + 1 < D_MODEL / 32) {
      issue((i + 1) * 32, buf ^ 1);
      AWAIT_ASYNC(0x2);
    } else {
      AWAIT_ASYNC(0x0);
    }
    __syncthreads();
    acc = wmma_bf16(frag_row(As[buf], 32, tr * 16, 0, lane),
                    frag_colT(BsT[buf], 32, 0, tc * 16, lane), acc);
    __syncthreads();
  }
  const int n = lane & 15, hh = lane >> 4;
#pragma unroll
  for (int r = 0; r < 8; ++r) {
    int row = row_blk + tr * 16 + r + 8 * hh;
    int col = col_blk + tc * 16 + n;
    out[(size_t)row * D_MODEL + col] = acc[r] + bias[col];
  }
}

// --- Kernel 4: LayerNorm ----------------------------------------------------
__global__ __launch_bounds__(256) void ln_kernel(const float* __restrict__ src,
                                                 const float* __restrict__ g,
                                                 const float* __restrict__ bta,
                                                 float* __restrict__ out) {
  __shared__ float red[256];
  __shared__ float red2[256];
  __shared__ float s_mu, s_rstd;
  int row = blockIdx.x, tid = threadIdx.x;
  float v0 = src[(size_t)row * D_MODEL + tid];
  float v1 = src[(size_t)row * D_MODEL + 256 + tid];
  red[tid] = v0 + v1;
  red2[tid] = v0 * v0 + v1 * v1;
  __syncthreads();
  for (int s = 128; s > 0; s >>= 1) {
    if (tid < s) { red[tid] += red[tid + s]; red2[tid] += red2[tid + s]; }
    __syncthreads();
  }
  if (tid == 0) {
    float mu = red[0] / (float)D_MODEL;
    float var = red2[0] / (float)D_MODEL - mu * mu;
    s_mu = mu;
    s_rstd = rsqrtf(var + 1e-5f);
  }
  __syncthreads();
  out[(size_t)row * D_MODEL + tid] = (v0 - s_mu) * s_rstd * g[tid] + bta[tid];
  out[(size_t)row * D_MODEL + 256 + tid] =
      (v1 - s_mu) * s_rstd * g[tid + 256] + bta[tid + 256];
}

// ---------------------------------------------------------------------------
extern "C" void kernel_launch(void* const* d_in, const int* in_sizes, int n_in,
                              void* d_out, int out_size, void* d_ws, size_t ws_size,
                              hipStream_t stream) {
  const float* x  = (const float*)d_in[0];
  const float* Wq = (const float*)d_in[1];
  const float* bq = (const float*)d_in[2];
  const float* Wo = (const float*)d_in[3];
  const float* bo = (const float*)d_in[4];
  const float* dl = (const float*)d_in[5];
  const float* lg = (const float*)d_in[6];
  const float* lb = (const float*)d_in[7];

  char* w = (char*)d_ws;
  bf16_t* xb    = (bf16_t*)w; w += (size_t)ROWS * D_MODEL * 2;
  bf16_t* wqbT  = (bf16_t*)w; w += (size_t)D_MODEL * N_QKV * 2;   // [1536][512]
  bf16_t* wobT  = (bf16_t*)w; w += (size_t)D_MODEL * D_MODEL * 2; // [512][512]
  bf16_t* qws   = (bf16_t*)w; w += (size_t)ROWS * D_MODEL * 2;
  bf16_t* kws   = (bf16_t*)w; w += (size_t)ROWS * D_MODEL * 2;
  bf16_t* vws   = (bf16_t*)w; w += (size_t)ROWS * D_MODEL * 2;
  bf16_t* attn  = (bf16_t*)w; w += (size_t)ROWS * D_MODEL * 2;
  float*  pre   = (float*)w;  w += (size_t)ROWS * D_MODEL * 4;
  // total workspace 30 MiB

  convert_kernel<<<(ROWS * D_MODEL) / 256, 256, 0, stream>>>(x, xb, ROWS * D_MODEL);
  convert_t_kernel<<<(D_MODEL * N_QKV) / 256, 256, 0, stream>>>(Wq, wqbT, D_MODEL, N_QKV);
  convert_t_kernel<<<(D_MODEL * D_MODEL) / 256, 256, 0, stream>>>(Wo, wobT, D_MODEL, D_MODEL);

  qkv_gemm_kernel<<<dim3(N_QKV / 32, ROWS / 64), 256, 0, stream>>>(
      xb, wqbT, bq, dl, qws, kws, vws);
  scan_kernel<<<BATCH * HEADS, 256, 0, stream>>>(qws, kws, vws, attn);
  out_gemm_kernel<<<dim3(D_MODEL / 32, ROWS / 64), 256, 0, stream>>>(
      attn, wobT, bo, pre);
  ln_kernel<<<ROWS, 256, 0, stream>>>(pre, lg, lb, (float*)d_out);
}
